// PfhedgeAttentionLSTM_48473000902909
// MI455X (gfx1250) — matrix operand order repaired and checked by reading, not compiled, not run on hardware
//
#include <hip/hip_runtime.h>

// Problem constants
#define TT   2048
#define BB   128
#define DD   4
#define HH   64
#define AA   32
#define WW   10
#define GG   256          // 4*H gate width
#define BT   16           // batch tile per workgroup
#define XCH  64           // x timesteps per TDM chunk
#define SCALE_INV 0.17677669529663687f   // 1/sqrt(32)
#define LOG2E     1.4426950408889634f

typedef __attribute__((ext_vector_type(16))) __bf16 v16bf;
typedef __attribute__((ext_vector_type(8)))  float  v8f;
typedef __attribute__((ext_vector_type(4)))  unsigned v4u;
typedef __attribute__((ext_vector_type(4)))  int      v4i;
typedef __attribute__((ext_vector_type(8)))  int      v8i;

union Frag { v16bf v; unsigned u[8]; };

__device__ __forceinline__ unsigned short f2bf(float f) {
  unsigned u = __float_as_uint(f);
  u += 0x7FFFu + ((u >> 16) & 1u);              // round-to-nearest-even
  return (unsigned short)(u >> 16);
}
__device__ __forceinline__ float bf2f(unsigned short b) {
  return __uint_as_float(((unsigned)b) << 16);
}

// ---- CDNA5 hardware transcendentals (branch-free) ----
__device__ __forceinline__ float fexp2(float x) {
#if __has_builtin(__builtin_amdgcn_exp2f)
  return __builtin_amdgcn_exp2f(x);     // raw v_exp_f32
#else
  return __expf(x * 0.6931471805599453f);
#endif
}
__device__ __forceinline__ float ftanh(float x) {
#if __has_builtin(__builtin_amdgcn_tanhf)
  return __builtin_amdgcn_tanhf(x);     // v_tanh_f32
#else
  float xc = fminf(fmaxf(x, -10.0f), 10.0f);
  float e  = fexp2(xc * (2.0f * LOG2E));
  return 1.0f - 2.0f / (e + 1.0f);
#endif
}
__device__ __forceinline__ float frcp(float x) {
#if __has_builtin(__builtin_amdgcn_rcpf)
  return __builtin_amdgcn_rcpf(x);      // v_rcp_f32
#else
  return 1.0f / x;
#endif
}
__device__ __forceinline__ float sigm(float x) {
  return 0.5f + 0.5f * ftanh(0.5f * x); // one TRANS op + FMA
}

// ---- Tensor Data Mover (async x staging) ----
#if __has_builtin(__builtin_amdgcn_tensor_load_to_lds)
#define HAVE_TDM 1
#else
#define HAVE_TDM 0
#endif

__device__ __forceinline__ void tdm_wait0() {
#if __has_builtin(__builtin_amdgcn_s_wait_tensorcnt)
  __builtin_amdgcn_s_wait_tensorcnt((short)0);
#else
  asm volatile("s_wait_tensorcnt 0x0" ::: "memory");
#endif
}

#if HAVE_TDM
// 2D tile: dim0 = XCH*DD contiguous fp32 per batch row, dim1 = BT rows,
// row stride = TT*DD elements. D# built per ISA §8.3/8.4 bit layout.
__device__ __forceinline__ void tdm_issue_x(const float* X, int b0, int t0,
                                            const float* lds_dst) {
  unsigned long long ga = (unsigned long long)(const void*)(X + (size_t)b0 * (TT * DD) + t0 * DD);
  unsigned lds_off = (unsigned)(unsigned long long)(const void*)lds_dst; // LDS generic addr[31:0]
  v4u g0;
  g0[0] = 1u;                                           // count=1, user mode
  g0[1] = lds_off;                                      // lds_addr (bytes)
  g0[2] = (unsigned)ga;                                 // global_addr[31:0]
  g0[3] = (unsigned)((ga >> 32) & 0x1FFFFFFu) | (2u << 30); // addr[56:32] | type=2
  v8i g1;
  g1[0] = (int)(2u << 16);                              // data_size=4B
  g1[1] = (int)(((unsigned)(TT * DD) & 0xFFFFu) << 16); // tensor_dim0[15:0]
  g1[2] = (int)(((unsigned)(TT * DD) >> 16) | ((unsigned)BT << 16)); // dim0 hi | tensor_dim1 lo
  g1[3] = (int)((unsigned)(XCH * DD) << 16);            // tile_dim0 = 256
  g1[4] = BT;                                           // tile_dim1 = 16
  g1[5] = TT * DD;                                      // tensor_dim0_stride lo32 = 8192
  g1[6] = 0;
  g1[7] = 0;
  v4i g2 = {0, 0, 0, 0};
  v4i g3 = {0, 0, 0, 0};
#if defined(__clang_major__) && __clang_major__ >= 23
  v8i g4 = {0, 0, 0, 0, 0, 0, 0, 0};
  __builtin_amdgcn_tensor_load_to_lds(g0, g1, g2, g3, g4, 0);
#else
  __builtin_amdgcn_tensor_load_to_lds(g0, g1, g2, g3, 0);
#endif
}
#endif

// B-matrix fragment (KxN tile, N=16, K=32) from row-major W[G][ldk] (fp32 global).
__device__ __forceinline__ v16bf make_bfrag(const float* W, int ldk, int n0, int k0, int lane) {
  int n  = lane & 15;
  int kh = (lane >> 4) << 4;
  const float* row = W + (n0 + n) * ldk;
  Frag f;
#pragma unroll
  for (int v = 0; v < 8; ++v) {
    int kb = k0 + ((v & 4) ? 16 : 0) + (v & 3) * 2 + (kh >> 1);
    f.u[v] = (unsigned)f2bf(row[kb]) | ((unsigned)f2bf(row[kb + 1]) << 16);
  }
  return f.v;
}

// A-matrix fragment (16x32) from bf16 LDS matrix h[16][64].
__device__ __forceinline__ v16bf make_afrag(const unsigned short* h, int k0, int lane) {
  int m  = lane & 15;
  int kh = (lane >> 4) * 8;
  const unsigned* row = (const unsigned*)(h + m * HH);
  Frag f;
#pragma unroll
  for (int v = 0; v < 8; ++v) {
    int kb = k0 + ((v & 4) ? 16 : 0) + (v & 3) * 2 + kh;
    f.u[v] = row[kb >> 1];
  }
  return f.v;
}

__device__ __forceinline__ v8f wmma_bf16(v16bf a, v16bf b, v8f c) {
  return __builtin_amdgcn_wmma_f32_16x16x32_bf16(false, a, false, b, (short)0, c, false, false);
}

__global__ __launch_bounds__(256, 1)
void PfhedgeAttentionLSTM_kernel(
    const float* __restrict__ X,
    const float* __restrict__ Wih0, const float* __restrict__ Whh0,
    const float* __restrict__ bih0, const float* __restrict__ bhh0,
    const float* __restrict__ Wih1, const float* __restrict__ Whh1,
    const float* __restrict__ bih1, const float* __restrict__ bhh1,
    const float* __restrict__ Wq,  const float* __restrict__ bq,
    const float* __restrict__ Wk,  const float* __restrict__ bk,
    const float* __restrict__ Wv,  const float* __restrict__ bv,
    const float* __restrict__ Wc,  const float* __restrict__ bc,
    const float* __restrict__ Wo1, const float* __restrict__ bo1,
    const float* __restrict__ Wo2, const float* __restrict__ bo2,
    float* __restrict__ out)
{
  // ---- LDS state ----
  __shared__ __align__(16) unsigned short h0s[BT * HH];    // layer0 h (bf16)
  __shared__ __align__(16) unsigned short h1s[BT * HH];    // layer1 h (bf16)
  __shared__ __align__(16) unsigned short ctxs[BT * HH];   // attention context (bf16)
  __shared__ __align__(16) unsigned short combs[BT * HH];  // combined (bf16)
  __shared__ float gsh[BT * GG];                           // gate pre-activations (fp32)
  __shared__ __align__(16) float xbuf[2][BT][XCH * DD];    // TDM double buffer (2x16KB)
  __shared__ float qs[BT * AA];                            // q (fp32)
  __shared__ __align__(16) unsigned short ks[WW][BT * AA]; // k ring (bf16)
  __shared__ __align__(16) unsigned short vs[WW][BT * HH]; // v ring (bf16)
  __shared__ float scs[BT * WW];                           // scores -> weights
  __shared__ float o1s[BT * AA];                           // relu MLP hidden
  __shared__ float wo2s[AA];

  const int tid  = threadIdx.x;
  const int lane = tid & 31;
  const int wv   = tid >> 5;            // wave id 0..7
  const int nn   = lane & 15;           // C/D column within tile
  const int mh   = (lane >> 4) * 8;     // C/D row base (0 or 8)
  const int b0   = blockIdx.x * BT;

  // ---- loop-invariant weight fragments kept in VGPRs (recurrence-critical only) ----
  v16bf Bhh0[2][2], Bih1[2][2], Bhh1[2][2];
  float bias0[2], bias1[2], wx[2][4];
#pragma unroll
  for (int ti = 0; ti < 2; ++ti) {
    int g0 = (2 * wv + ti) * 16;
#pragma unroll
    for (int kc = 0; kc < 2; ++kc) {
      Bhh0[ti][kc] = make_bfrag(Whh0, HH, g0, kc * 32, lane);
      Bih1[ti][kc] = make_bfrag(Wih1, HH, g0, kc * 32, lane);
      Bhh1[ti][kc] = make_bfrag(Whh1, HH, g0, kc * 32, lane);
    }
    int g = g0 + nn;
    bias0[ti] = bih0[g] + bhh0[g];
    bias1[ti] = bih1[g] + bhh1[g];
#pragma unroll
    for (int d = 0; d < DD; ++d) wx[ti][d] = Wih0[g * DD + d];
  }

  // qkv projection: waves 0-1 -> q tiles, 2-3 -> k tiles, 4-7 -> v tiles
  const float* qkvW; const float* qkvB; int qkv_n0;
  if (wv < 2)       { qkvW = Wq; qkvB = bq; qkv_n0 = wv * 16; }
  else if (wv < 4)  { qkvW = Wk; qkvB = bk; qkv_n0 = (wv - 2) * 16; }
  else              { qkvW = Wv; qkvB = bv; qkv_n0 = (wv - 4) * 16; }
  v16bf Bqkv[2];
  Bqkv[0] = make_bfrag(qkvW, HH, qkv_n0, 0,  lane);
  Bqkv[1] = make_bfrag(qkvW, HH, qkv_n0, 32, lane);
  const float biasqkv = qkvB[qkv_n0 + nn];

  // Wc (waves 0-3) / Wo1 (waves 4-5): fragments rebuilt per step from global.
  const float biasc  = (wv < 4) ? bc[wv * 16 + nn] : 0.0f;
  const float biaso1 = (wv == 4 || wv == 5) ? bo1[(wv - 4) * 16 + nn] : 0.0f;
  const float bo2v = bo2[0];
  if (tid < AA) wo2s[tid] = Wo2[tid];

  // ---- init state ----
  float c0r[4], c1r[4];
#pragma unroll
  for (int p = 0; p < 4; ++p) { c0r[p] = 0.0f; c1r[p] = 0.0f; }
  for (int i = tid; i < BT * HH; i += 256) { h0s[i] = 0; h1s[i] = 0; }
  __syncthreads();

  // ================== time recurrence ==================
  for (int t = 0; t < TT; ++t) {
    const int slot = t % WW;
    const int xb   = (t >> 6) & 1;
    const int tin  = t & (XCH - 1);

    // Launder post-LSTM weight pointers (LICM blocker).
    const float* wcp  = Wc;
    const float* wo1p = Wo1;
    asm volatile("" : "+v"(wcp), "+v"(wo1p));

    // (0) chunk boundary: retire current TDM transfer, kick off the next one
    if (tin == 0) {
#if HAVE_TDM
      if (wv == 0) {
        if (t == 0) tdm_issue_x(X, b0, 0, &xbuf[0][0][0]);   // first chunk
        tdm_wait0();                                         // current chunk resident
        if (t + XCH < TT)
          tdm_issue_x(X, b0, t + XCH, &xbuf[xb ^ 1][0][0]);  // prefetch next chunk
      }
#else
      for (int i = tid; i < BT * XCH * DD; i += 256) {
        int row = i >> 8, col = i & (XCH * DD - 1);
        xbuf[xb][row][col] = X[(b0 + row) * (TT * DD) + t * DD + col];
      }
#endif
      __syncthreads();
    }

    // (1) layer0 gates: g0 = x@Wih0^T + h0@Whh0^T + b   (h0 = previous step)
    {
      v16bf a0 = make_afrag(h0s, 0, lane);
      v16bf a1 = make_afrag(h0s, 32, lane);
#pragma unroll
      for (int ti = 0; ti < 2; ++ti) {
        v8f acc;
#pragma unroll
        for (int r = 0; r < 8; ++r) {
          const float* xr = &xbuf[xb][mh + r][tin * DD];   // aligned 16B -> ds_load_b128
          acc[r] = bias0[ti] + xr[0] * wx[ti][0] + xr[1] * wx[ti][1]
                             + xr[2] * wx[ti][2] + xr[3] * wx[ti][3];
        }
        acc = wmma_bf16(a0, Bhh0[ti][0], acc);
        acc = wmma_bf16(a1, Bhh0[ti][1], acc);
#pragma unroll
        for (int r = 0; r < 8; ++r)
          gsh[(mh + r) * GG + (2 * wv + ti) * 16 + nn] = acc[r];
      }
    }
    __syncthreads();

    // (2) layer0 elementwise update: each lane owns 4 (m, nh) cells
#pragma unroll
    for (int p = 0; p < 4; ++p) {
      int q = tid * 4 + p, m = q >> 6, nh = q & 63;
      const float* gr = &gsh[m * GG];
      float gi = sigm(gr[nh]);
      float gf = sigm(gr[HH + nh]);
      float gg = ftanh(gr[2 * HH + nh]);
      float go = sigm(gr[3 * HH + nh]);
      c0r[p] = gf * c0r[p] + gi * gg;
      h0s[q] = f2bf(go * ftanh(c0r[p]));
    }
    __syncthreads();

    // (3) layer1 gates: g1 = h0_new@Wih1^T + h1_old@Whh1^T + b
    {
      v16bf ax0 = make_afrag(h0s, 0, lane);
      v16bf ax1 = make_afrag(h0s, 32, lane);
      v16bf ah0 = make_afrag(h1s, 0, lane);
      v16bf ah1 = make_afrag(h1s, 32, lane);
#pragma unroll
      for (int ti = 0; ti < 2; ++ti) {
        v8f acc;
#pragma unroll
        for (int r = 0; r < 8; ++r) acc[r] = bias1[ti];
        acc = wmma_bf16(ax0, Bih1[ti][0], acc);
        acc = wmma_bf16(ax1, Bih1[ti][1], acc);
        acc = wmma_bf16(ah0, Bhh1[ti][0], acc);
        acc = wmma_bf16(ah1, Bhh1[ti][1], acc);
#pragma unroll
        for (int r = 0; r < 8; ++r)
          gsh[(mh + r) * GG + (2 * wv + ti) * 16 + nn] = acc[r];
      }
    }
    __syncthreads();

    // (4) layer1 elementwise update
#pragma unroll
    for (int p = 0; p < 4; ++p) {
      int q = tid * 4 + p, m = q >> 6, nh = q & 63;
      const float* gr = &gsh[m * GG];
      float gi = sigm(gr[nh]);
      float gf = sigm(gr[HH + nh]);
      float gg = ftanh(gr[2 * HH + nh]);
      float go = sigm(gr[3 * HH + nh]);
      c1r[p] = gf * c1r[p] + gi * gg;
      h1s[q] = f2bf(go * ftanh(c1r[p]));
    }
    __syncthreads();

    // (5) q/k/v projections from h1_new
    {
      v16bf a0 = make_afrag(h1s, 0, lane);
      v16bf a1 = make_afrag(h1s, 32, lane);
      v8f acc;
#pragma unroll
      for (int r = 0; r < 8; ++r) acc[r] = biasqkv;
      acc = wmma_bf16(a0, Bqkv[0], acc);
      acc = wmma_bf16(a1, Bqkv[1], acc);
      if (wv < 2) {
#pragma unroll
        for (int r = 0; r < 8; ++r) qs[(mh + r) * AA + qkv_n0 + nn] = acc[r];
      } else if (wv < 4) {
#pragma unroll
        for (int r = 0; r < 8; ++r) ks[slot][(mh + r) * AA + qkv_n0 + nn] = f2bf(acc[r]);
      } else {
#pragma unroll
        for (int r = 0; r < 8; ++r) vs[slot][(mh + r) * HH + qkv_n0 + nn] = f2bf(acc[r]);
      }
    }
    __syncthreads();

    if (t > 0) {
      // (6a) scores over the W=10 causal window
      if (tid < BT * WW) {
        int m = tid / WW, w = tid % WW;
        float s = -1.0e9f;
        if (w < t) {
          int sl = (t - 1 - w) % WW;
          const unsigned short* kk = &ks[sl][m * AA];
          const float* qq = &qs[m * AA];
          float d = 0.0f;
#pragma unroll
          for (int j = 0; j < AA; ++j) d += qq[j] * bf2f(kk[j]);
          s = d * SCALE_INV;
        }
        scs[m * WW + w] = s;
      }
      __syncthreads();
      // (6b) softmax per row (raw v_exp_f32: args <= 0 after max-subtract)
      if (tid < BT) {
        float mx = -1.0e30f;
#pragma unroll
        for (int w = 0; w < WW; ++w) mx = fmaxf(mx, scs[tid * WW + w]);
        float sum = 0.0f;
#pragma unroll
        for (int w = 0; w < WW; ++w) {
          float e = fexp2((scs[tid * WW + w] - mx) * LOG2E);
          scs[tid * WW + w] = e; sum += e;
        }
        float inv = frcp(sum);
#pragma unroll
        for (int w = 0; w < WW; ++w) scs[tid * WW + w] *= inv;
      }
      __syncthreads();
      // (6c) context = weights @ v_window
      {
        int wlim = (t < WW) ? t : WW;
#pragma unroll
        for (int p = 0; p < 4; ++p) {
          int q = tid * 4 + p, m = q >> 6, hh = q & 63;
          float cacc = 0.0f;
          for (int w = 0; w < wlim; ++w) {
            int sl = (t - 1 - w) % WW;
            cacc += scs[m * WW + w] * bf2f(vs[sl][m * HH + hh]);
          }
          ctxs[q] = f2bf(cacc);
        }
      }
      __syncthreads();
      // (6d) combined = [h1, ctx] @ Wc^T + bc    (waves 0-3, K=128)
      if (wv < 4) {
        v16bf ah0 = make_afrag(h1s, 0, lane);
        v16bf ah1 = make_afrag(h1s, 32, lane);
        v16bf ac0 = make_afrag(ctxs, 0, lane);
        v16bf ac1 = make_afrag(ctxs, 32, lane);
        v8f acc;
#pragma unroll
        for (int r = 0; r < 8; ++r) acc[r] = biasc;
        acc = wmma_bf16(ah0, make_bfrag(wcp, 2 * HH, wv * 16, 0,  lane), acc);
        acc = wmma_bf16(ah1, make_bfrag(wcp, 2 * HH, wv * 16, 32, lane), acc);
        acc = wmma_bf16(ac0, make_bfrag(wcp, 2 * HH, wv * 16, 64, lane), acc);
        acc = wmma_bf16(ac1, make_bfrag(wcp, 2 * HH, wv * 16, 96, lane), acc);
#pragma unroll
        for (int r = 0; r < 8; ++r)
          combs[(mh + r) * HH + wv * 16 + nn] = f2bf(acc[r]);
      }
      __syncthreads();
    } else {
      // t == 0: combined := lstm_out
      for (int i = tid; i < BT * HH; i += 256) combs[i] = h1s[i];
      __syncthreads();
    }

    // (7) MLP hidden: relu(combined @ Wo1^T + bo1)   (waves 4-5)
    if (wv == 4 || wv == 5) {
      int n0 = (wv - 4) * 16;
      v16bf a0 = make_afrag(combs, 0, lane);
      v16bf a1 = make_afrag(combs, 32, lane);
      v8f acc;
#pragma unroll
      for (int r = 0; r < 8; ++r) acc[r] = biaso1;
      acc = wmma_bf16(a0, make_bfrag(wo1p, HH, n0, 0,  lane), acc);
      acc = wmma_bf16(a1, make_bfrag(wo1p, HH, n0, 32, lane), acc);
#pragma unroll
      for (int r = 0; r < 8; ++r)
        o1s[(mh + r) * AA + n0 + nn] = fmaxf(acc[r], 0.0f);
    }
    __syncthreads();

    // (8) scalar head + output
    if (tid < BT) {
      float raw = bo2v;
      const float* orow = &o1s[tid * AA];
#pragma unroll
      for (int j = 0; j < AA; ++j) raw += orow[j] * wo2s[j];
      out[(b0 + tid) * TT + t] = 1.5f * ftanh(raw);
    }
    __syncthreads();
  }
}

extern "C" void kernel_launch(void* const* d_in, const int* in_sizes, int n_in,
                              void* d_out, int out_size, void* d_ws, size_t ws_size,
                              hipStream_t stream) {
  (void)in_sizes; (void)n_in; (void)d_ws; (void)ws_size; (void)out_size;
  const float* X    = (const float*)d_in[0];
  const float* Wih0 = (const float*)d_in[1];
  const float* Whh0 = (const float*)d_in[2];
  const float* bih0 = (const float*)d_in[3];
  const float* bhh0 = (const float*)d_in[4];
  const float* Wih1 = (const float*)d_in[5];
  const float* Whh1 = (const float*)d_in[6];
  const float* bih1 = (const float*)d_in[7];
  const float* bhh1 = (const float*)d_in[8];
  const float* Wq   = (const float*)d_in[9];
  const float* bq   = (const float*)d_in[10];
  const float* Wk   = (const float*)d_in[11];
  const float* bk   = (const float*)d_in[12];
  const float* Wv   = (const float*)d_in[13];
  const float* bv   = (const float*)d_in[14];
  const float* Wc   = (const float*)d_in[15];
  const float* bc   = (const float*)d_in[16];
  const float* Wo1  = (const float*)d_in[17];
  const float* bo1  = (const float*)d_in[18];
  const float* Wo2  = (const float*)d_in[19];
  const float* bo2  = (const float*)d_in[20];
  float* out = (float*)d_out;

  dim3 grid(BB / BT);   // 8 workgroups, one 16-row batch tile each
  dim3 block(256);      // 8 wave32 waves
  PfhedgeAttentionLSTM_kernel<<<grid, block, 0, stream>>>(
      X, Wih0, Whh0, bih0, bhh0, Wih1, Whh1, bih1, bhh1,
      Wq, bq, Wk, bk, Wv, bv, Wc, bc, Wo1, bo1, Wo2, bo2, out);
}
